// DirectedHyperConvLayer_34136400068849
// MI455X (gfx1250) — matrix operand order repaired and checked by reading, not compiled, not run on hardware
//
#include <hip/hip_runtime.h>
#include <hip/hip_bf16.h>

#define ALPHA_T 0.1f
#define EPS_V   1e-8f
#define D       64

typedef __attribute__((ext_vector_type(16))) _Float16 v16h;
typedef __attribute__((ext_vector_type(8)))  float    v8f;
typedef __attribute__((ext_vector_type(2)))  _Float16 h2;

// ---------------------------------------------------------------------------
// A-matrix fragment (16x32 f16, M x K) from an LDS tile of 16 rows x 64 halves.
// ISA layout: lanes 0-15 -> M=lane, K-base 0 ; lanes 16-31 -> M=lane-16, K-base 8.
// VGPR j (j=0..3): K = kb+2j, kb+2j+1 ; VGPR 4+j: K = kb+16+2j, kb+16+2j+1.
// ---------------------------------------------------------------------------
__device__ __forceinline__ v16h ldsFragA(const _Float16* base, int lane, int c0) {
    const int m  = lane & 15;
    const int kb = (lane < 16) ? 0 : 8;
    const _Float16* row = base + m * D + c0 + kb;
    v16h f;
#pragma unroll
    for (int j = 0; j < 4; ++j) {
        h2 p0 = *(const h2*)(row + 2 * j);        // K = kb+2j, kb+2j+1
        h2 p1 = *(const h2*)(row + 16 + 2 * j);   // K = kb+16+2j, +1
        f[2 * j]     = p0[0]; f[2 * j + 1]     = p0[1];
        f[8 + 2 * j] = p1[0]; f[8 + 2 * j + 1] = p1[1];
    }
    return f;
}

// ---------------------------------------------------------------------------
// B-matrix fragment (32x16 f16, K x N) = X^T for an LDS tile (row n along K).
// Layout: lanes 0-15 -> K-base 0 ; lanes 16-31 -> K-base 8 ; VGPR j<4: K=kb+2j,+1 ; VGPR j>=4: K=kb+16+2(j-4),+1.
// ---------------------------------------------------------------------------
__device__ __forceinline__ v16h ldsFragB(const _Float16* base, int lane, int c0) {
    const int n  = lane & 15;
    const int kb = (lane < 16) ? 0 : 8;
    const _Float16* row = base + n * D + c0 + kb;
    v16h f;
#pragma unroll
    for (int j = 0; j < 8; ++j) {
        h2 p = *(const h2*)(row + 2 * j + (j >= 4 ? 8 : 0));
        f[2 * j] = p[0]; f[2 * j + 1] = p[1];
    }
    return f;
}

// ---------------------------------------------------------------------------
// Refine kernel: one wave32 handles 16 edges. Gathers src/tar rows to LDS
// (f32 -> f16), then computes diag(S.T^T), diag(S.S^T), diag(T.T^T) with
// v_wmma_f32_16x16x32_f16 and forms refined edge weights in f32.
// ---------------------------------------------------------------------------
__global__ __launch_bounds__(256)
void refine_vals_wmma_kernel(const float* __restrict__ embs,
                             const int*   __restrict__ eidx,   // [2, E]
                             const float* __restrict__ base_val,
                             float*       __restrict__ refined,
                             int E) {
    __shared__ __align__(16) _Float16 lds[8][2][16][D];   // 32 KB / block

    const int lane = threadIdx.x & 31;
    const int wave = threadIdx.x >> 5;
    const int tile = blockIdx.x * 8 + wave;
    const int e0   = tile * 16;

    _Float16 (*S)[D] = lds[wave][0];
    _Float16 (*T)[D] = lds[wave][1];

    // Gather phase: per edge, 64 f32 per row, 2 per lane, coalesced 256B.
#pragma unroll 1
    for (int e = 0; e < 16; ++e) {
        int ee = e0 + e;
        if (ee >= E) ee = E - 1;              // wave-uniform clamp (tail)
        const long long rs = eidx[ee];
        const long long rt = eidx[E + ee];
        float2 vs = *(const float2*)(embs + rs * D + 2 * lane);
        float2 vt = *(const float2*)(embs + rt * D + 2 * lane);
        h2 hs; hs[0] = (_Float16)vs.x; hs[1] = (_Float16)vs.y;
        h2 ht; ht[0] = (_Float16)vt.x; ht[1] = (_Float16)vt.y;
        *(h2*)&S[e][2 * lane] = hs;
        *(h2*)&T[e][2 * lane] = ht;
    }
    __syncthreads();

    // Gram matrices over K=64 (two K=32 chunks). EXEC is all ones here.
    v8f cDot = {}; v8f cNS = {}; v8f cNT = {};
#pragma unroll
    for (int c0 = 0; c0 < D; c0 += 32) {
        v16h aS = ldsFragA(&S[0][0], lane, c0);
        v16h aT = ldsFragA(&T[0][0], lane, c0);
        v16h bS = ldsFragB(&S[0][0], lane, c0);
        v16h bT = ldsFragB(&T[0][0], lane, c0);
        cDot = __builtin_amdgcn_wmma_f32_16x16x32_f16(false, aS, false, bT,
                                                      (short)0, cDot, false, false);
        cNS  = __builtin_amdgcn_wmma_f32_16x16x32_f16(false, aS, false, bS,
                                                      (short)0, cNS, false, false);
        cNT  = __builtin_amdgcn_wmma_f32_16x16x32_f16(false, aT, false, bT,
                                                      (short)0, cNT, false, false);
    }

    // Diagonal extraction: (m,m) is at VGPR m / lane m (m<8) or
    // VGPR m-8 / lane m+16 (m>=8). Compile-time selects avoid dynamic VGPR idx.
    float dDot = 0.f, dNS = 0.f, dNT = 0.f;
#pragma unroll
    for (int r = 0; r < 8; ++r) {
        const bool sel = (lane == r) || (lane == r + 24);
        dDot = sel ? cDot[r] : dDot;
        dNS  = sel ? cNS[r]  : dNS;
        dNT  = sel ? cNT[r]  : dNT;
    }
    const int srcLane = ((lane < 8) ? lane : lane + 16) & 31;  // lanes 0..15 valid
    const float dot = __shfl(dDot, srcLane, 32);
    const float ns  = __shfl(dNS,  srcLane, 32);
    const float nt  = __shfl(dNT,  srcLane, 32);

    if (lane < 16) {
        const int ee = e0 + lane;
        if (ee < E) {
            const float n1  = fmaxf(sqrtf(fmaxf(ns, 0.f)), EPS_V);
            const float n2  = fmaxf(sqrtf(fmaxf(nt, 0.f)), EPS_V);
            const float sim = dot / (n1 * n2);
            const float sn  = fminf(fmaxf((sim + 1.f) * 0.5f, 0.f), 1.f);
            refined[ee] = base_val[ee] * (1.f + ALPHA_T * sn);
        }
    }
}

// ---------------------------------------------------------------------------
// COO SpMM via L2 atomics: one wave per edge, 2 f32 per lane.
// out[row, :] += val * x[col, :]
// ---------------------------------------------------------------------------
__global__ __launch_bounds__(256)
void spmm_atomic_kernel(const int*   __restrict__ eidx,  // [2, E]
                        const float* __restrict__ vals,
                        const float* __restrict__ x,
                        float*       __restrict__ out,
                        int E) {
    const int gid  = blockIdx.x * blockDim.x + threadIdx.x;
    const int lane = gid & 31;
    const int edge = gid >> 5;
    if (edge >= E) return;
    const long long row = eidx[edge];
    const long long col = eidx[E + edge];
    const float v = vals[edge];
    const float2 xv = *(const float2*)(x + col * D + 2 * lane);
    float* o = out + row * D + 2 * lane;
    atomicAdd(o + 0, v * xv.x);
    atomicAdd(o + 1, v * xv.y);
}

// ---------------------------------------------------------------------------
extern "C" void kernel_launch(void* const* d_in, const int* in_sizes, int n_in,
                              void* d_out, int out_size, void* d_ws, size_t ws_size,
                              hipStream_t stream) {
    const float* embs    = (const float*)d_in[0];
    const int*   src_idx = (const int*)d_in[1];   // harness delivers ints as int32
    const float* src_val = (const float*)d_in[2];
    const int*   tar_idx = (const int*)d_in[3];
    const float* tar_val = (const float*)d_in[4];

    const int E = in_sizes[2];          // 800000
    const int N = in_sizes[0] / D;      // 50000
    float* out = (float*)d_out;

    float* refined_src = (float*)d_ws;
    float* refined_tar = refined_src + E;
    float* msg_tar     = refined_tar + E;

    // Accumulators must start at zero every call (atomic scatter).
    hipMemsetAsync(msg_tar, 0, (size_t)N * D * sizeof(float), stream);
    hipMemsetAsync(out,     0, (size_t)out_size * sizeof(float), stream);

    const int tiles   = (E + 15) / 16;
    const int rblocks = (tiles + 7) / 8;          // 8 waves / block
    refine_vals_wmma_kernel<<<rblocks, 256, 0, stream>>>(embs, tar_idx, tar_val,
                                                         refined_tar, E);
    refine_vals_wmma_kernel<<<rblocks, 256, 0, stream>>>(embs, src_idx, src_val,
                                                         refined_src, E);

    const long long sthreads = (long long)E * 32;
    const int sblocks = (int)((sthreads + 255) / 256);
    // msg_tar = spmm(tar_edges, refined_tar, embs)
    spmm_atomic_kernel<<<sblocks, 256, 0, stream>>>(tar_idx, refined_tar, embs,
                                                    msg_tar, E);
    // out = spmm(src_edges, refined_src, msg_tar)
    spmm_atomic_kernel<<<sblocks, 256, 0, stream>>>(src_idx, refined_src, msg_tar,
                                                    out, E);
}
